// FlashMultiHeadAttention_82437602279902
// MI455X (gfx1250) — matrix operand order repaired and checked by reading, not compiled, not run on hardware
//
#include <hip/hip_runtime.h>
#include <hip/hip_bf16.h>

// ---------------------------------------------------------------------------
// Problem constants (match reference)
// ---------------------------------------------------------------------------
#define H_    1024
#define NH_   16
#define HD_   64
#define B_    32
#define S_    500
#define SP_   512            // S padded to a multiple of 32 for WMMA tiling
#define MTOT_ (B_ * S_)      // 16000 GEMM rows (multiple of 16)

typedef __attribute__((ext_vector_type(16))) _Float16 v16h;
typedef __attribute__((ext_vector_type(8)))  float    v8f;

// D = A(16x32 f16) * B(32x16 f16) + C(16x16 f32), wave32 CDNA5 WMMA
__device__ __forceinline__ v8f wmma_f16(v16h a, v16h b, v8f c) {
  return __builtin_amdgcn_wmma_f32_16x16x32_f16(
      /*neg_a=*/false, a, /*neg_b=*/false, b,
      /*c_mod=*/(short)0, c, /*reuse_a=*/false, /*reuse_b=*/false);
}

// ---------------------------------------------------------------------------
// Zero-fill (padding rows of Q/K/V head buffers must be 0, ws is poisoned)
// ---------------------------------------------------------------------------
__global__ void zero_half_kernel(_Float16* __restrict__ p, long n) {
  long stride = (long)gridDim.x * blockDim.x * 8;
  for (long i = ((long)blockIdx.x * blockDim.x + threadIdx.x) * 8; i < n; i += stride) {
    *reinterpret_cast<float4*>(p + i) = make_float4(0.f, 0.f, 0.f, 0.f);
  }
}

// ---------------------------------------------------------------------------
// GEMM: out = X @ W^T + bias.   X:[16000,1024]  W:[1024,1024] (row-major)
// Block: 256 thr = 8 waves. Wave -> 16x64 tile. Grid: (125 Mtiles, 16 Ntiles).
// N-tile of 64 == one head (HD_==64), so SPLIT writes [B,NH,SP,HD] f16.
// ---------------------------------------------------------------------------
template <bool HALF_IN, bool SPLIT>
__global__ void __launch_bounds__(256)
gemm_kernel(const void* __restrict__ Xv, const float* __restrict__ W,
            const float* __restrict__ bias, void* __restrict__ outv) {
  __shared__ _Float16 Wl[32][HD_ + 4];   // k-slab of W^T: [k][n], f16

  const int tid  = threadIdx.x;
  const int wave = tid >> 5;
  const int lane = tid & 31;
  const int ln16 = lane & 15;
  const int hi   = lane >> 4;            // half-wave select (ISA A/C layouts)
  const int mtile = blockIdx.x;
  const int nbase = blockIdx.y * HD_;
  const int mrow  = mtile * 128 + wave * 16 + ln16;   // A-matrix row of this lane
  const int g     = hi ? 8 : 0;

  v8f c0 = {}, c1 = {}, c2 = {}, c3 = {};

  for (int kt = 0; kt < H_ / 32; ++kt) {
    const int kbase = kt * 32;
    __syncthreads();
    {   // cooperative stage: W[n, k] (fp32) -> Wl[k][n] (f16), 8 elems/thread
      const int n  = tid >> 2;
      const int k0 = (tid & 3) * 8;
      const float* ws = W + (size_t)(nbase + n) * H_ + kbase + k0;
#pragma unroll
      for (int j = 0; j < 8; ++j) Wl[k0 + j][n] = (_Float16)ws[j];
    }
    __syncthreads();

    // A fragment (16x32): lane ln16 = row M; e<8 -> K=g+e, e>=8 -> K=16+g+(e-8)
    v16h a;
    if (HALF_IN) {
      const _Float16* row = (const _Float16*)Xv + (size_t)mrow * H_ + kbase;
#pragma unroll
      for (int e = 0; e < 8; ++e) { a[e] = row[g + e]; a[8 + e] = row[16 + g + e]; }
    } else {
      const float* row = (const float*)Xv + (size_t)mrow * H_ + kbase;
#pragma unroll
      for (int e = 0; e < 8; ++e) {
        a[e]     = (_Float16)row[g + e];
        a[8 + e] = (_Float16)row[16 + g + e];
      }
    }

    // B fragments (32x16): lane ln16 = N; e<8 -> K=g+e, e>=8 -> K=16+g+(e-8)
    v16h b0, b1, b2, b3;
#pragma unroll
    for (int e = 0; e < 8; ++e) {
      b0[e] = Wl[g + e][ln16];      b0[8 + e] = Wl[16 + g + e][ln16];
      b1[e] = Wl[g + e][16 + ln16]; b1[8 + e] = Wl[16 + g + e][16 + ln16];
      b2[e] = Wl[g + e][32 + ln16]; b2[8 + e] = Wl[16 + g + e][32 + ln16];
      b3[e] = Wl[g + e][48 + ln16]; b3[8 + e] = Wl[16 + g + e][48 + ln16];
    }
    c0 = wmma_f16(a, b0, c0);
    c1 = wmma_f16(a, b1, c1);
    c2 = wmma_f16(a, b2, c2);
    c3 = wmma_f16(a, b3, c3);
  }

  const float bia0 = bias[nbase + ln16];
  const float bia1 = bias[nbase + 16 + ln16];
  const float bia2 = bias[nbase + 32 + ln16];
  const float bia3 = bias[nbase + 48 + ln16];

  // C/D layout: VGPR r -> M = r (lanes 0-15) or r+8 (lanes 16-31); N = ln16
#pragma unroll
  for (int r = 0; r < 8; ++r) {
    const int m = mtile * 128 + wave * 16 + (hi ? r + 8 : r);
    const float v0 = c0[r] + bia0, v1 = c1[r] + bia1;
    const float v2 = c2[r] + bia2, v3 = c3[r] + bia3;
    if (SPLIT) {
      const int bb = m / S_, s = m - bb * S_;
      const int hh = blockIdx.y;   // N-tile == head index
      _Float16* dst =
          (_Float16*)outv + (((size_t)bb * NH_ + hh) * SP_ + s) * HD_ + ln16;
      dst[0]  = (_Float16)v0;
      dst[16] = (_Float16)v1;
      dst[32] = (_Float16)v2;
      dst[48] = (_Float16)v3;
    } else {
      float* dst = (float*)outv + (size_t)m * H_ + nbase + ln16;
      dst[0] = v0; dst[16] = v1; dst[32] = v2; dst[48] = v3;
    }
  }
}

// ---------------------------------------------------------------------------
// RoPE (even/odd interleave variant), in place on [B*NH, SP, HD] f16.
// One wave handles one (b,h,s) row; lane j covers dims j and j+32.
// ---------------------------------------------------------------------------
__global__ void __launch_bounds__(256) rope_kernel(_Float16* __restrict__ X) {
  const int tid   = threadIdx.x;
  const int lane  = tid & 31;
  const int rowid = blockIdx.x * 8 + (tid >> 5);  // [0, B*NH*SP)
  const int s     = rowid & (SP_ - 1);
  _Float16* row   = X + (size_t)rowid * HD_;

  const float inv_freq = __powf(10000.0f, -(float)(2 * lane) / (float)HD_);
  const float f  = (float)s * inv_freq;
  const float cs = __cosf(f), sn = __sinf(f);

  const float x0 = (float)row[lane];
  const float x1 = (float)row[lane + 32];
  const float e0 = (float)row[2 * lane];
  const float e1 = (float)row[2 * lane + 1];
  __syncthreads();  // all reads before in-place writes (row lives in one wave)
  row[lane]      = (_Float16)(x0 * cs - e1 * sn);  // x_rot[j]    = -x[2j+1]
  row[lane + 32] = (_Float16)(x1 * cs + e0 * sn);  // x_rot[j+32] =  x[2j]
}

// ---------------------------------------------------------------------------
// Flash attention. Block = one (b,h) x 128 q-rows (8 waves x 16 rows).
// kv loop in 32-wide tiles; K staged transposed, V natural, P bounced via LDS.
// Bias/mask turned into per-block LDS LUTs so the hot loop is WMMA + ds + exp.
// ---------------------------------------------------------------------------
__global__ void __launch_bounds__(256)
attn_kernel(const _Float16* __restrict__ Qh, const _Float16* __restrict__ Kh,
            const _Float16* __restrict__ Vh, const float* __restrict__ rel_emb,
            const unsigned char* __restrict__ amask, _Float16* __restrict__ Oh) {
  __shared__ _Float16 Kt[HD_][34];       // K^T tile: [d][kv_local]
  __shared__ _Float16 Vt[32][HD_ + 4];   // V tile:   [kv_local][d]
  __shared__ _Float16 Pt[8][16][34];     // per-wave P: [q_local][kv_local]
  __shared__ float    rel_lds[1024];     // bias(delta = q-kv), delta+512
  __shared__ float    msk_lds[SP_];      // additive key mask (0 / -1e30)

  const int tid  = threadIdx.x;
  const int wave = tid >> 5;
  const int lane = tid & 31;
  const int ln16 = lane & 15;
  const int hi   = lane >> 4;
  const int bh = blockIdx.x;
  const int b  = bh >> 4;
  const int h  = bh & 15;
  const int qbase = blockIdx.y * 128 + wave * 16;

  const _Float16* Qb = Qh + (size_t)bh * SP_ * HD_;
  const _Float16* Kb = Kh + (size_t)bh * SP_ * HD_;
  const _Float16* Vb = Vh + (size_t)bh * SP_ * HD_;

  // ---- build per-block LUTs (h and b are fixed for the whole block) ----
#pragma unroll
  for (int i = tid; i < 1024; i += 256) {
    int idx = (i - 512) + (S_ - 1);                       // delta + 499
    idx = idx < 0 ? 0 : (idx > 2 * S_ - 2 ? 2 * S_ - 2 : idx);
    rel_lds[i] = rel_emb[idx * NH_ + h];
  }
#pragma unroll
  for (int i = tid; i < SP_; i += 256) {
    const bool ok = (i < S_) && amask[b * S_ + i];
    msk_lds[i] = ok ? 0.0f : -1e30f;
  }

  // Q A-fragments, persistent across kv loop (d split 0-31 / 32-63)
  v16h aq0, aq1;
  {
    const _Float16* qrow = Qb + (size_t)(qbase + ln16) * HD_;
    const int g = hi ? 8 : 0;
#pragma unroll
    for (int e = 0; e < 8; ++e) {
      aq0[e]     = qrow[g + e];
      aq0[8 + e] = qrow[16 + g + e];
      aq1[e]     = qrow[32 + g + e];
      aq1[8 + e] = qrow[48 + g + e];
    }
  }

  float m_i[8], l_i[8];
  v8f o0 = {}, o1 = {}, o2 = {}, o3 = {};
#pragma unroll
  for (int r = 0; r < 8; ++r) { m_i[r] = -1e30f; l_i[r] = 0.0f; }

  __syncthreads();   // LUTs ready

  for (int kt = 0; kt < SP_ / 32; ++kt) {
    const int kvbase = kt * 32;
    {   // cooperative stage of K (transposed) and V tiles, 8 halves/thread
      const int row = tid >> 3;
      const int dc  = (tid & 7) * 8;
      const _Float16* ks = Kb + (size_t)(kvbase + row) * HD_ + dc;
      const _Float16* vs = Vb + (size_t)(kvbase + row) * HD_ + dc;
      if (kt + 1 < SP_ / 32) {            // global_prefetch_b8 next tiles
        __builtin_prefetch(ks + 32 * HD_, 0, 1);
        __builtin_prefetch(vs + 32 * HD_, 0, 1);
      }
#pragma unroll
      for (int j = 0; j < 8; ++j) {
        Kt[dc + j][row] = ks[j];
        Vt[row][dc + j] = vs[j];
      }
    }
    __syncthreads();

    // scores: S(16x32) = Q(16x64) @ K^T(64x32), 4 WMMAs
    v8f s0 = {}, s1 = {};
    {
      v16h b00, b01, b10, b11;   // [d-step][kv-group]
      const int gk = hi ? 8 : 0;
#pragma unroll
      for (int e = 0; e < 8; ++e) {
        b00[e] = Kt[gk + e][ln16];           b00[8 + e] = Kt[16 + gk + e][ln16];
        b01[e] = Kt[gk + e][16 + ln16];      b01[8 + e] = Kt[16 + gk + e][16 + ln16];
        b10[e] = Kt[32 + gk + e][ln16];      b10[8 + e] = Kt[48 + gk + e][ln16];
        b11[e] = Kt[32 + gk + e][16 + ln16]; b11[8 + e] = Kt[48 + gk + e][16 + ln16];
      }
      s0 = wmma_f16(aq0, b00, s0);
      s0 = wmma_f16(aq1, b10, s0);
      s1 = wmma_f16(aq0, b01, s1);
      s1 = wmma_f16(aq1, b11, s1);
    }

    // scale + rel-pos bias + key mask (pure LDS adds) + online softmax
    const int   kv0 = kvbase + ln16;
    const int   kv1 = kv0 + 16;
    const float mb0 = msk_lds[kv0];
    const float mb1 = msk_lds[kv1];
    // per-lane base into rel LUT; row r adds +1, fragment 1 adds -16
    const float* relb = rel_lds + (qbase + (hi ? 8 : 0) - kv0 + 512);
    const float scale = 0.125f;   // 1/sqrt(64)
#pragma unroll
    for (int r = 0; r < 8; ++r) {
      float v0 = fmaf(s0[r], scale, relb[r]      + mb0);
      float v1 = fmaf(s1[r], scale, relb[r - 16] + mb1);

      float mx = fmaxf(v0, v1);
#pragma unroll
      for (int off = 8; off >= 1; off >>= 1) mx = fmaxf(mx, __shfl_xor(mx, off, 32));
      const float mnew = fmaxf(m_i[r], mx);
      const float corr = __expf(m_i[r] - mnew);
      m_i[r] = mnew;
      o0[r] *= corr; o1[r] *= corr; o2[r] *= corr; o3[r] *= corr;
      const float p0 = __expf(v0 - mnew);
      const float p1 = __expf(v1 - mnew);
      float sm = p0 + p1;
#pragma unroll
      for (int off = 8; off >= 1; off >>= 1) sm += __shfl_xor(sm, off, 32);
      l_i[r] = l_i[r] * corr + sm;

      const int ql = hi ? r + 8 : r;     // C-layout row -> LDS [q][kv]
      Pt[wave][ql][ln16]      = (_Float16)p0;
      Pt[wave][ql][16 + ln16] = (_Float16)p1;
    }
    __syncthreads();   // P visible (also fences Kt/Vt reuse; uniform barriers)

    // O(16x64) += P(16x32) @ V(32x64), 4 WMMAs
    {
      v16h ap;
      const _Float16* prow = Pt[wave][ln16];
      const int g = hi ? 8 : 0;
#pragma unroll
      for (int e = 0; e < 8; ++e) { ap[e] = prow[g + e]; ap[8 + e] = prow[16 + g + e]; }
      v16h bv0, bv1, bv2, bv3;
#pragma unroll
      for (int e = 0; e < 8; ++e) {
        bv0[e] = Vt[g + e][ln16];      bv0[8 + e] = Vt[16 + g + e][ln16];
        bv1[e] = Vt[g + e][16 + ln16]; bv1[8 + e] = Vt[16 + g + e][16 + ln16];
        bv2[e] = Vt[g + e][32 + ln16]; bv2[8 + e] = Vt[16 + g + e][32 + ln16];
        bv3[e] = Vt[g + e][48 + ln16]; bv3[8 + e] = Vt[16 + g + e][48 + ln16];
      }
      o0 = wmma_f16(ap, bv0, o0);
      o1 = wmma_f16(ap, bv1, o1);
      o2 = wmma_f16(ap, bv2, o2);
      o3 = wmma_f16(ap, bv3, o3);
    }
    __syncthreads();   // all waves done reading tiles before restage
  }

  // epilogue: normalize, clip to +-30, write f16 [B,S,H] for the O-projection
#pragma unroll
  for (int r = 0; r < 8; ++r) {
    const int q = qbase + (hi ? r + 8 : r);
    if (q < S_) {
      const float inv = 1.0f / l_i[r];
      _Float16* dst = Oh + ((size_t)b * S_ + q) * H_ + h * HD_ + ln16;
      dst[0]  = (_Float16)fminf(30.f, fmaxf(-30.f, o0[r] * inv));
      dst[16] = (_Float16)fminf(30.f, fmaxf(-30.f, o1[r] * inv));
      dst[32] = (_Float16)fminf(30.f, fmaxf(-30.f, o2[r] * inv));
      dst[48] = (_Float16)fminf(30.f, fmaxf(-30.f, o3[r] * inv));
    }
  }
}

// ---------------------------------------------------------------------------
// Launch
// ---------------------------------------------------------------------------
extern "C" void kernel_launch(void* const* d_in, const int* in_sizes, int n_in,
                              void* d_out, int out_size, void* d_ws, size_t ws_size,
                              hipStream_t stream) {
  const float* q  = (const float*)d_in[0];
  const float* k  = (const float*)d_in[1];
  const float* v  = (const float*)d_in[2];
  const unsigned char* amask = (const unsigned char*)d_in[3];  // jax bool = 1B
  const float* Wq = (const float*)d_in[4];
  const float* bq = (const float*)d_in[5];
  const float* Wk = (const float*)d_in[6];
  const float* bk = (const float*)d_in[7];
  const float* Wv = (const float*)d_in[8];
  const float* bv = (const float*)d_in[9];
  const float* Wo = (const float*)d_in[10];
  const float* bo = (const float*)d_in[11];
  const float* rel = (const float*)d_in[12];

  // workspace: Qh|Kh|Vh f16 [B,NH,SP,HD] (32 MiB each) + Oh f16 [B,S,H]
  const size_t HEADBUF = (size_t)B_ * NH_ * SP_ * HD_;
  _Float16* Qh = (_Float16*)d_ws;
  _Float16* Kh = Qh + HEADBUF;
  _Float16* Vh = Kh + HEADBUF;
  _Float16* Oh = Vh + HEADBUF;

  zero_half_kernel<<<2048, 256, 0, stream>>>(Qh, (long)(3 * HEADBUF));

  const dim3 gg(MTOT_ / 128, H_ / HD_);   // (125, 16)
  gemm_kernel<false, true><<<gg, 256, 0, stream>>>((const void*)q, Wq, bq, (void*)Qh);
  gemm_kernel<false, true><<<gg, 256, 0, stream>>>((const void*)k, Wk, bk, (void*)Kh);
  gemm_kernel<false, true><<<gg, 256, 0, stream>>>((const void*)v, Wv, bv, (void*)Vh);

  rope_kernel<<<(B_ * NH_ * SP_) / 8, 256, 0, stream>>>(Qh);
  rope_kernel<<<(B_ * NH_ * SP_) / 8, 256, 0, stream>>>(Kh);

  attn_kernel<<<dim3(B_ * NH_, SP_ / 128), 256, 0, stream>>>(Qh, Kh, Vh, rel, amask, Oh);

  gemm_kernel<true, false><<<gg, 256, 0, stream>>>((const void*)Oh, Wo, bo, d_out);
}
